// LoRALinear_6554120093993
// MI455X (gfx1250) — matrix operand order repaired and checked by reading, not compiled, not run on hardware
//
#include <hip/hip_runtime.h>

typedef __attribute__((ext_vector_type(16))) _Float16 v16h;
typedef __attribute__((ext_vector_type(8)))  _Float16 v8h;
typedef __attribute__((ext_vector_type(4)))  _Float16 v4h;
typedef __attribute__((ext_vector_type(8)))  float    v8f;
typedef __attribute__((ext_vector_type(4)))  float    v4f;
typedef __attribute__((ext_vector_type(4)))  int      v4i;

#define IN_DIM    4096
#define OUT_DIM   4096
#define RANK      16
#define M_TOTAL   8192          // 4 * 2048 tokens
#define NF4_BLOCK 64
#define LORA_SCALE 2.0f         // ALPHA / RANK = 32 / 16

// NF4 codebook (QLoRA)
__constant__ float NF4_TAB[16] = {
    -1.0f, -0.6961928009986877f, -0.5250730514526367f, -0.39491748809814453f,
    -0.28444138169288635f, -0.18477343022823334f, -0.09105003625154495f, 0.0f,
    0.07958029955625534f, 0.16093020141124725f, 0.24611230194568634f,
    0.33791524171829224f, 0.44070982933044434f, 0.5626170039176941f,
    0.7229568362236328f, 1.0f};

// ---------------------------------------------------------------------------
// Kernel 0: convert x (f32) -> f16, 8 elements per thread
// ---------------------------------------------------------------------------
__global__ __launch_bounds__(256) void cvt_x_f16(const float* __restrict__ x,
                                                 _Float16* __restrict__ xh) {
  size_t i = ((size_t)blockIdx.x * 256 + threadIdx.x) * 8;
  v4f a = *(const v4f*)(x + i);
  v4f b = *(const v4f*)(x + i + 4);
  v8h h;
#pragma unroll
  for (int t = 0; t < 4; ++t) {
    h[t]     = (_Float16)a[t];
    h[t + 4] = (_Float16)b[t];
  }
  *(v8h*)(xh + i) = h;
}

// ---------------------------------------------------------------------------
// Kernel 1: W_eff[o][i] = NF4[code]*scale + 2 * (lora_b[o,:] . lora_a[:,i])
// stored f16, row-major [OUT_DIM][IN_DIM]. 4 consecutive i per thread.
// ---------------------------------------------------------------------------
__global__ __launch_bounds__(256) void build_weff(
    const int* __restrict__ codes, const float* __restrict__ scales,
    const float* __restrict__ la, const float* __restrict__ lb,
    _Float16* __restrict__ wh) {
  int idx = blockIdx.x * 256 + threadIdx.x;      // OUT_DIM*IN_DIM/4 threads
  int o  = idx >> 10;                            // IN_DIM/4 = 1024 per row
  int i4 = (idx & 1023) << 2;

  float s = scales[o * (IN_DIM / NF4_BLOCK) + (i4 >> 6)];
  v4i c = *(const v4i*)(codes + (size_t)o * IN_DIM + i4);

  float acc0 = 0.f, acc1 = 0.f, acc2 = 0.f, acc3 = 0.f;
  const float* lbo = lb + o * RANK;
#pragma unroll
  for (int r = 0; r < RANK; ++r) {
    float bo = lbo[r];
    v4f av = *(const v4f*)(la + (size_t)r * IN_DIM + i4);
    acc0 += bo * av[0];
    acc1 += bo * av[1];
    acc2 += bo * av[2];
    acc3 += bo * av[3];
  }

  v4h w;
  w[0] = (_Float16)(NF4_TAB[c[0]] * s + LORA_SCALE * acc0);
  w[1] = (_Float16)(NF4_TAB[c[1]] * s + LORA_SCALE * acc1);
  w[2] = (_Float16)(NF4_TAB[c[2]] * s + LORA_SCALE * acc2);
  w[3] = (_Float16)(NF4_TAB[c[3]] * s + LORA_SCALE * acc3);
  *(v4h*)(wh + (size_t)o * IN_DIM + i4) = w;
}

// ---------------------------------------------------------------------------
// Kernel 2: out[m][n] = sum_k xh[m][k] * wh[n][k], WMMA f32_16x16x32_f16
// Block tile 128x128, BK=32, 256 threads = 8 wave32 (wave grid 4M x 2N,
// each wave computes 2x4 16x16 tiles = 32x64 of C).
// ---------------------------------------------------------------------------
#define BM 128
#define BN 128
#define BK 32
#define LDSW 40   // padded row stride in halves (32 data + 8 pad)

__global__ __launch_bounds__(256) void gemm_wmma(
    const _Float16* __restrict__ xh, const _Float16* __restrict__ wh,
    float* __restrict__ out) {
  __shared__ _Float16 Xs[BM * LDSW];
  __shared__ _Float16 Ws[BN * LDSW];

  const int tid  = threadIdx.x;
  const int wave = tid >> 5;
  const int lane = tid & 31;
  const int lr   = lane & 15;   // row within 16x16 fragment
  const int lg   = lane >> 4;   // K-group select

  const int m0 = blockIdx.y * BM;
  const int n0 = blockIdx.x * BN;
  const int wm = (wave >> 1) * 32;   // wave M offset in block: 0,32,64,96
  const int wn = (wave & 1) * 64;    // wave N offset in block: 0,64

  // staging: thread t loads rows (t>>2) and (t>>2)+64, 8 halves at col (t&3)*8
  const int srow = tid >> 2;
  const int scol = (tid & 3) * 8;

  const _Float16* xg = xh + (size_t)(m0 + srow) * IN_DIM + scol;
  const _Float16* wg = wh + (size_t)(n0 + srow) * IN_DIM + scol;

  v8f acc[2][4] = {};

  // software pipeline: preload first K-tile into registers
  v8h rx0 = *(const v8h*)(xg);
  v8h rx1 = *(const v8h*)(xg + (size_t)64 * IN_DIM);
  v8h rw0 = *(const v8h*)(wg);
  v8h rw1 = *(const v8h*)(wg + (size_t)64 * IN_DIM);

  for (int k0 = 0; k0 < IN_DIM; k0 += BK) {
    __syncthreads();               // previous iteration's LDS reads complete
    *(v8h*)&Xs[srow * LDSW + scol]        = rx0;
    *(v8h*)&Xs[(srow + 64) * LDSW + scol] = rx1;
    *(v8h*)&Ws[srow * LDSW + scol]        = rw0;
    *(v8h*)&Ws[(srow + 64) * LDSW + scol] = rw1;
    __syncthreads();

    const int k1 = k0 + BK;
    if (k1 < IN_DIM) {             // prefetch next K-tile into registers
      rx0 = *(const v8h*)(xg + k1);
      rx1 = *(const v8h*)(xg + (size_t)64 * IN_DIM + k1);
      rw0 = *(const v8h*)(wg + k1);
      rw1 = *(const v8h*)(wg + (size_t)64 * IN_DIM + k1);
    }
    if (k0 + 2 * BK < IN_DIM) {    // L2 prefetch two tiles ahead
      __builtin_prefetch(xg + k0 + 2 * BK, 0, 3);
      __builtin_prefetch(wg + k0 + 2 * BK, 0, 3);
    }

    // A fragments: 16-bit A 16x32 layout -> lane holds row lr, K chunks at
    // [lg*8, lg*8+8) and [16+lg*8, 16+lg*8+8)
    v16h a[2], b[4];
#pragma unroll
    for (int i = 0; i < 2; ++i) {
      const _Float16* p = &Xs[(wm + i * 16 + lr) * LDSW];
      v8h lo = *(const v8h*)(p + lg * 8);
      v8h hi = *(const v8h*)(p + 16 + lg * 8);
      a[i] = __builtin_shufflevector(lo, hi, 0, 1, 2, 3, 4, 5, 6, 7, 8, 9, 10,
                                     11, 12, 13, 14, 15);
    }
    // B fragments: B 32x16 layout -> lane holds column lr, K chunks at
    // [lg*8, lg*8+8) and [16+lg*8, 16+lg*8+8); Ws is [n][k]
#pragma unroll
    for (int j = 0; j < 4; ++j) {
      const _Float16* p = &Ws[(wn + j * 16 + lr) * LDSW];
      v8h lo = *(const v8h*)(p + lg * 8);
      v8h hi = *(const v8h*)(p + 16 + lg * 8);
      b[j] = __builtin_shufflevector(lo, hi, 0, 1, 2, 3, 4, 5, 6, 7, 8, 9, 10,
                                     11, 12, 13, 14, 15);
    }

#pragma unroll
    for (int i = 0; i < 2; ++i)
#pragma unroll
      for (int j = 0; j < 4; ++j)
        acc[i][j] = __builtin_amdgcn_wmma_f32_16x16x32_f16(
            false, a[i], false, b[j], (short)0, acc[i][j], false, false);
  }

  // Epilogue: D layout -> lane holds col (lane&15), VGPR r is row r+8*(lane>>4)
#pragma unroll
  for (int i = 0; i < 2; ++i) {
#pragma unroll
    for (int j = 0; j < 4; ++j) {
      const int n     = n0 + wn + j * 16 + lr;
      const int mbase = m0 + wm + i * 16 + lg * 8;
#pragma unroll
      for (int r = 0; r < 8; ++r)
        out[(size_t)(mbase + r) * OUT_DIM + n] = acc[i][j][r];
    }
  }
}

// ---------------------------------------------------------------------------
extern "C" void kernel_launch(void* const* d_in, const int* in_sizes, int n_in,
                              void* d_out, int out_size, void* d_ws,
                              size_t ws_size, hipStream_t stream) {
  const float* x      = (const float*)d_in[0];
  const int*   codes  = (const int*)d_in[1];
  const float* scales = (const float*)d_in[2];
  const float* la     = (const float*)d_in[3];
  const float* lb     = (const float*)d_in[4];
  float*       out    = (float*)d_out;

  _Float16* xh = (_Float16*)d_ws;                                  // 64 MB
  _Float16* wh = (_Float16*)((char*)d_ws +
                             (size_t)M_TOTAL * IN_DIM * sizeof(_Float16));  // +32 MB

  const int cvt_blocks = (int)(((size_t)M_TOTAL * IN_DIM) / (256 * 8));   // 16384
  cvt_x_f16<<<cvt_blocks, 256, 0, stream>>>(x, xh);

  const int w_blocks = (int)(((size_t)OUT_DIM * IN_DIM) / (256 * 4));     // 16384
  build_weff<<<w_blocks, 256, 0, stream>>>(codes, scales, la, lb, wh);

  dim3 grid(OUT_DIM / BN, M_TOTAL / BM);   // (32, 64)
  gemm_wmma<<<grid, 256, 0, stream>>>(xh, wh, out);
}
